// LSHAttention_31903017075353
// MI455X (gfx1250) — matrix operand order repaired and checked by reading, not compiled, not run on hardware
//
#include <hip/hip_runtime.h>
#include <hip/hip_bf16.h>
#include <math.h>

typedef __attribute__((ext_vector_type(16))) _Float16 v16h;
typedef __attribute__((ext_vector_type(8)))  _Float16 v8h;
typedef __attribute__((ext_vector_type(8)))  float    v8f;

#define BATCH  4
#define SEQ    8192
#define DHEAD  64
#define NHASH  8
#define BSIZE  64
#define NBUCK  128                 // buckets per hash
#define TOTB   (NHASH*NBUCK)       // 1024 distinct bucket ids (encodes hash)
#define TOTAL  (NHASH*SEQ)         // 65536 tickers per batch
#define NCHUNK (TOTAL/BSIZE)       // 1024 chunks per batch

// ---------------------------------------------------------------------------
// 1) LSH hashing: rotated = qk @ rot  (per hash), bucket = argmax([r, -r])
// ---------------------------------------------------------------------------
__global__ __launch_bounds__(256) void lsh_hash_kernel(
    const float* __restrict__ qk, const float* __restrict__ rot,
    int* __restrict__ buckets, float* __restrict__ bucketsF)
{
  __shared__ float rl[DHEAD * 64];
  int tid = threadIdx.x;
  int gid = blockIdx.x * 256 + tid;          // 0 .. B*S-1
  int b = gid >> 13;
  int t = gid & (SEQ - 1);

  float x[DHEAD];
  const float4* qr = (const float4*)(qk + ((size_t)b * SEQ + t) * DHEAD);
#pragma unroll
  for (int i = 0; i < 16; i++) {
    float4 f = qr[i];
    x[4*i+0] = f.x; x[4*i+1] = f.y; x[4*i+2] = f.z; x[4*i+3] = f.w;
  }

  for (int h = 0; h < NHASH; h++) {
    for (int i = tid; i < DHEAD * 64; i += 256) {
      int d = i >> 6, j = i & 63;
      rl[i] = rot[(d * NHASH + h) * 64 + j];   // rotations (1,D,H,64)
    }
    __syncthreads();

    float pmax = -3.0e38f, nmax = -3.0e38f;
    int   pidx = 0,        nidx = 0;
    for (int j = 0; j < 64; j++) {
      float acc = 0.f;
#pragma unroll
      for (int d = 0; d < DHEAD; d++) acc += x[d] * rl[d * 64 + j];
      if (acc > pmax) { pmax = acc; pidx = j; }     // first-max tie-break
      float na = -acc;
      if (na > nmax)  { nmax = na;  nidx = j; }
    }
    // concat order [r, -r]: any positive-side index precedes negative side
    int bucket = (pmax >= nmax) ? pidx : (nidx + 64);
    bucket += h * NBUCK;
    int oi = b * TOTAL + h * SEQ + t;
    buckets[oi]  = bucket;
    bucketsF[oi] = (float)bucket;   // second output of the reference tuple
    __syncthreads();
  }
}

// ---------------------------------------------------------------------------
// 2) Per-batch histogram of bucket ids
// ---------------------------------------------------------------------------
__global__ __launch_bounds__(256) void hist_kernel(
    const int* __restrict__ buckets, int* __restrict__ hist)
{
  int idx = blockIdx.x * 256 + threadIdx.x;   // B*TOTAL threads exactly
  int b = idx >> 16;
  atomicAdd(&hist[b * TOTB + buckets[idx]], 1);
}

// ---------------------------------------------------------------------------
// 3) Exclusive scan of 1024 bucket counts (one block per batch)
// ---------------------------------------------------------------------------
__global__ __launch_bounds__(1024) void scan_kernel(
    const int* __restrict__ hist, int* __restrict__ base)
{
  __shared__ int sd[TOTB];
  int b = blockIdx.x, tid = threadIdx.x;
  int v = hist[b * TOTB + tid];
  sd[tid] = v;
  __syncthreads();
  for (int off = 1; off < TOTB; off <<= 1) {
    int t = (tid >= off) ? sd[tid - off] : 0;
    __syncthreads();
    sd[tid] += t;
    __syncthreads();
  }
  base[b * TOTB + tid] = sd[tid] - v;         // exclusive
}

// ---------------------------------------------------------------------------
// 4) Stable gather-sort: block = (bucket, batch); ballot-ranked scatter.
//    Result: sticker[b][j] = ticker at sorted position j (== jnp.argsort).
// ---------------------------------------------------------------------------
__global__ __launch_bounds__(256) void sort_kernel(
    const int* __restrict__ buckets, const int* __restrict__ base,
    int* __restrict__ sticker)
{
  int q = blockIdx.x, b = blockIdx.y;
  const int* bk = buckets + b * TOTAL;
  int dst = base[b * TOTB + q];
  __shared__ int wsum[8];
  int lane = threadIdx.x & 31, wv = threadIdx.x >> 5;
  for (int start = 0; start < TOTAL; start += 256) {
    int i = start + threadIdx.x;
    bool pred = (bk[i] == q);
    unsigned mask = (unsigned)__ballot(pred ? 1 : 0);   // wave32: low 32 bits
    if (lane == 0) wsum[wv] = __popc(mask);
    __syncthreads();
    int wpre = 0, ctotal = 0;
#pragma unroll
    for (int k = 0; k < 8; k++) { int c = wsum[k]; ctotal += c; if (k < wv) wpre += c; }
    if (pred) {
      int rank = wpre + __popc(mask & ((1u << lane) - 1u));
      sticker[b * TOTAL + dst + rank] = i;
    }
    dst += ctotal;
    __syncthreads();
  }
}

// ---------------------------------------------------------------------------
// 5) Chunk attention with WMMA f16: one 128-thread block per (chunk, batch).
//    GEMM1: q(64x64) @ k^T(64x128) ; mask(bq_t==bkv_t) ; softmax ;
//    GEMM2: p(64x128) @ v(128x64)  ; scatter rows + lse by ticker.
// ---------------------------------------------------------------------------
__global__ __launch_bounds__(128) void attn_kernel(
    const float* __restrict__ qk, const float* __restrict__ v,
    const int* __restrict__ sticker,
    float* __restrict__ logits_tmp, _Float16* __restrict__ o_tmp)
{
  constexpr int QS = 72;    // q/k LDS row stride (halves), 144B: 16B aligned, bank-spread
  constexpr int PS = 136;   // probs/vT LDS row stride (halves), 272B
  __shared__ __align__(16) _Float16 q_lds[64 * QS];
  __shared__ __align__(16) _Float16 k_lds[128 * QS];
  __shared__ __align__(16) _Float16 vT_lds[64 * PS];   // vT[d][key]
  __shared__ __align__(16) _Float16 p_lds[64 * PS];
  __shared__ int kvpos[128];   // bkv_t ; first 64 == bq_t
  __shared__ int tickc[64];    // ticker of current-chunk rows (for unsort scatter)

  int cid = blockIdx.x, b = blockIdx.y;
  int tid = threadIdx.x;
  int pid = (cid + NCHUNK - 1) & (NCHUNK - 1);          // look_one_back
  int j = (tid < 64) ? (cid * BSIZE + tid) : (pid * BSIZE + (tid - 64));
  int tick = sticker[b * TOTAL + j];
  int st = tick & (SEQ - 1);
  kvpos[tid] = st;
  if (tid < 64) tickc[tid] = tick;

  const float4* qr = (const float4*)(qk + ((size_t)b * SEQ + st) * DHEAD);
  const float4* vr = (const float4*)(v  + ((size_t)b * SEQ + st) * DHEAD);
  __builtin_prefetch(vr, 0, 1);                         // global_prefetch_b8

  float x[DHEAD];
  float ss = 0.f;
#pragma unroll
  for (int i = 0; i < 16; i++) {
    float4 f = qr[i];
    x[4*i+0] = f.x; x[4*i+1] = f.y; x[4*i+2] = f.z; x[4*i+3] = f.w;
    ss += f.x*f.x + f.y*f.y + f.z*f.z + f.w*f.w;
  }
  float ksc = 1.f / (sqrtf(ss) + 1e-6f);                // bk = bqk/(||bqk||+1e-6)
#pragma unroll
  for (int d = 0; d < DHEAD; d++) k_lds[tid * QS + d] = (_Float16)(x[d] * ksc);
  if (tid < 64) {
#pragma unroll
    for (int d = 0; d < DHEAD; d++) q_lds[tid * QS + d] = (_Float16)x[d];
  }
#pragma unroll
  for (int i = 0; i < 16; i++) {
    float4 f = vr[i];
    vT_lds[(4*i+0) * PS + tid] = (_Float16)f.x;
    vT_lds[(4*i+1) * PS + tid] = (_Float16)f.y;
    vT_lds[(4*i+2) * PS + tid] = (_Float16)f.z;
    vT_lds[(4*i+3) * PS + tid] = (_Float16)f.w;
  }
  __syncthreads();

  int w = tid >> 5, lane = tid & 31;
  int hf = lane >> 4, ln = lane & 15;
  union HV { v16h v; v8h h[2]; };

  // ---- GEMM1: A = 16 q rows (M = 16w+ln), ksteps K=0..31 / 32..63 ----
  HV a0, a1;
  {
    const _Float16* ap = q_lds + (16 * w + ln) * QS;
    a0.h[0] = *(const v8h*)(ap +  8 * hf);
    a0.h[1] = *(const v8h*)(ap + 16 + 8 * hf);
    a1.h[0] = *(const v8h*)(ap + 32 + 8 * hf);
    a1.h[1] = *(const v8h*)(ap + 48 + 8 * hf);
  }
  v8f acc[8];
  const v8f vzero = {0.f,0.f,0.f,0.f,0.f,0.f,0.f,0.f};
#pragma unroll
  for (int n = 0; n < 8; n++) {
    acc[n] = vzero;
    const _Float16* bp = k_lds + (16 * n + ln) * QS;    // B[K][N] = k[N][K]
    HV b0, b1;
    b0.h[0] = *(const v8h*)(bp + 8 * hf);
    b0.h[1] = *(const v8h*)(bp + 16 + 8 * hf);
    b1.h[0] = *(const v8h*)(bp + 32 + 8 * hf);
    b1.h[1] = *(const v8h*)(bp + 48 + 8 * hf);
    acc[n] = __builtin_amdgcn_wmma_f32_16x16x32_f16(false, a0.v, false, b0.v,
                                                    (short)0, acc[n], false, false);
    acc[n] = __builtin_amdgcn_wmma_f32_16x16x32_f16(false, a1.v, false, b1.v,
                                                    (short)0, acc[n], false, false);
  }

  // ---- scale, self-mask (keep where bq_t==bkv_t else -1e5), softmax ----
  const float dscale = 0.125f;   // 64^-0.5
  int kp[8];
#pragma unroll
  for (int n = 0; n < 8; n++) kp[n] = kvpos[16 * n + ln];
#pragma unroll
  for (int r = 0; r < 8; r++) {
    int m = 16 * w + 8 * hf + r;          // C layout: M = r + 8*(lane>=16)
    int qp = kvpos[m];
    float mx = -3.0e38f;
#pragma unroll
    for (int n = 0; n < 8; n++) {
      float dv = acc[n][r] * dscale;
      dv = (kp[n] == qp) ? dv : -1e5f;
      acc[n][r] = dv;
      mx = fmaxf(mx, dv);
    }
#pragma unroll
    for (int d = 1; d < 16; d <<= 1) mx = fmaxf(mx, __shfl_xor(mx, d, 32));
    float sum = 0.f;
#pragma unroll
    for (int n = 0; n < 8; n++) {
      float e = __expf(acc[n][r] - mx);
      acc[n][r] = e;
      sum += e;
    }
#pragma unroll
    for (int d = 1; d < 16; d <<= 1) sum += __shfl_xor(sum, d, 32);
    float inv = 1.f / sum;
#pragma unroll
    for (int n = 0; n < 8; n++)
      p_lds[m * PS + 16 * n + ln] = (_Float16)(acc[n][r] * inv);
    if (ln == 0)
      logits_tmp[(size_t)b * TOTAL + tickc[m]] = mx + __logf(sum);  // unsorted lse
  }
  __syncthreads();

  // ---- GEMM2: bo = probs(64x128) @ v(128x64), 4 ksteps of 32 ----
  v8f oacc[4];
#pragma unroll
  for (int n = 0; n < 4; n++) oacc[n] = vzero;
#pragma unroll
  for (int ks = 0; ks < 4; ks++) {
    HV a;
    const _Float16* ap = p_lds + (16 * w + ln) * PS + 32 * ks;
    a.h[0] = *(const v8h*)(ap +  8 * hf);
    a.h[1] = *(const v8h*)(ap + 16 + 8 * hf);
#pragma unroll
    for (int n = 0; n < 4; n++) {
      HV bb;
      const _Float16* bp = vT_lds + (16 * n + ln) * PS + 32 * ks + 8 * hf;
      bb.h[0] = *(const v8h*)(bp);
      bb.h[1] = *(const v8h*)(bp + 16);
      oacc[n] = __builtin_amdgcn_wmma_f32_16x16x32_f16(false, a.v, false, bb.v,
                                                       (short)0, oacc[n], false, false);
    }
  }

  // ---- scatter rows to unsorted ticker positions (== undo_sort) ----
#pragma unroll
  for (int r = 0; r < 8; r++) {
    int m = 16 * w + 8 * hf + r;
    size_t ob = ((size_t)b * TOTAL + (size_t)tickc[m]) * DHEAD;
#pragma unroll
    for (int n = 0; n < 4; n++)
      o_tmp[ob + 16 * n + ln] = (_Float16)oacc[n][r];
  }
}

// ---------------------------------------------------------------------------
// 6) Combine hash rounds with logit-softmax weights
// ---------------------------------------------------------------------------
__global__ __launch_bounds__(256) void combine_kernel(
    const float* __restrict__ logits_tmp, const _Float16* __restrict__ o_tmp,
    float* __restrict__ out)
{
  int idx = blockIdx.x * 256 + threadIdx.x;   // B*S threads exactly
  int b = idx >> 13, t = idx & (SEQ - 1);
  float l[NHASH], mx = -3.0e38f;
#pragma unroll
  for (int h = 0; h < NHASH; h++) {
    l[h] = logits_tmp[(size_t)b * TOTAL + h * SEQ + t];
    mx = fmaxf(mx, l[h]);
  }
  float wsum = 0.f;
#pragma unroll
  for (int h = 0; h < NHASH; h++) { l[h] = __expf(l[h] - mx); wsum += l[h]; }
  float inv = 1.f / wsum;
  float acc[DHEAD];
#pragma unroll
  for (int d = 0; d < DHEAD; d++) acc[d] = 0.f;
  for (int h = 0; h < NHASH; h++) {
    const _Float16* op = o_tmp + ((size_t)b * TOTAL + h * SEQ + t) * DHEAD;
    float wh = l[h] * inv;
#pragma unroll
    for (int d = 0; d < DHEAD; d++) acc[d] += wh * (float)op[d];
  }
  float* orow = out + ((size_t)b * SEQ + t) * DHEAD;
#pragma unroll
  for (int d = 0; d < DHEAD; d++) orow[d] = acc[d];
}

// ---------------------------------------------------------------------------
extern "C" void kernel_launch(void* const* d_in, const int* in_sizes, int n_in,
                              void* d_out, int out_size, void* d_ws, size_t ws_size,
                              hipStream_t stream)
{
  (void)in_sizes; (void)n_in; (void)out_size; (void)ws_size;
  const float* qk  = (const float*)d_in[0];
  const float* vv  = (const float*)d_in[1];
  const float* rot = (const float*)d_in[2];
  float* out = (float*)d_out;
  float* bucketsF = out + (size_t)BATCH * SEQ * DHEAD;   // 2nd tuple output

  char* ws = (char*)d_ws;
  int*      buckets_i = (int*)ws;                                   // 1 MB
  int*      sticker   = (int*)(ws + (1u << 20));                    // 1 MB
  int*      hist      = (int*)(ws + (2u << 20));                    // 16 KB
  int*      basep     = (int*)(ws + (2u << 20) + (1u << 16));       // 16 KB
  float*    logits    = (float*)(ws + (2u << 20) + (2u << 16));     // 1 MB
  _Float16* o_tmp     = (_Float16*)(ws + (2u << 20) + (2u << 16) + (1u << 20)); // 32 MB

  hipMemsetAsync(hist, 0, BATCH * TOTB * sizeof(int), stream);
  lsh_hash_kernel<<<dim3((BATCH * SEQ) / 256), 256, 0, stream>>>(qk, rot, buckets_i, bucketsF);
  hist_kernel<<<dim3((BATCH * TOTAL) / 256), 256, 0, stream>>>(buckets_i, hist);
  scan_kernel<<<dim3(BATCH), TOTB, 0, stream>>>(hist, basep);
  sort_kernel<<<dim3(TOTB, BATCH), 256, 0, stream>>>(buckets_i, basep, sticker);
  attn_kernel<<<dim3(NCHUNK, BATCH), 128, 0, stream>>>(qk, vv, sticker, logits, o_tmp);
  combine_kernel<<<dim3((BATCH * SEQ) / 256), 256, 0, stream>>>(logits, o_tmp, out);
}
